// MultiPPIMI_59004260712696
// MI455X (gfx1250) — matrix operand order repaired and checked by reading, not compiled, not run on hardware
//
#include <hip/hip_runtime.h>

// ---------------------------------------------------------------------------
// CDNA5 WMMA types / helpers
// ---------------------------------------------------------------------------
typedef __attribute__((ext_vector_type(16))) __bf16 v16bf_t;
typedef __attribute__((ext_vector_type(8)))  float  v8f_t;

__device__ __forceinline__ unsigned short f2bf(float f) {
  __bf16 h = (__bf16)f;                      // hardware cvt, RNE
  return __builtin_bit_cast(unsigned short, h);
}
__device__ __forceinline__ float bf2f(unsigned short h) {
  return __uint_as_float(((unsigned int)h) << 16);
}
__device__ __forceinline__ unsigned int pack2(float lo, float hi) {
  return (unsigned int)f2bf(lo) | ((unsigned int)f2bf(hi) << 16);
}

// Async global->LDS copy (CDNA5 VGLOBAL async op, tracked by ASYNCcnt)
__device__ __forceinline__ void async_copy_b128(unsigned int lds_off,
                                                const void* gaddr) {
  asm volatile("global_load_async_to_lds_b128 %0, %1, off"
               :: "v"(lds_off), "v"(gaddr) : "memory");
}
template <int N> __device__ __forceinline__ void wait_async() {
#if __has_builtin(__builtin_amdgcn_s_wait_asynccnt)
  __builtin_amdgcn_s_wait_asynccnt(N);
#else
  asm volatile("s_wait_asynccnt %0" :: "n"(N) : "memory");
#endif
}
__device__ __forceinline__ unsigned int lds_off_of(const void* p) {
  return (unsigned int)(unsigned long long)p;   // low 32 bits = LDS address
}

// ---------------------------------------------------------------------------
// All-bf16 tiled WMMA GEMM:  C = act( A[M,Kd] * B[N,Kd]^T + bias )
// Block tile 64(M) x 128(N), BK=32, 256 threads = 8 waves (2x4),
// wave tile 32x32 -> 4x v_wmma_f32_16x16x32_bf16 per k-step.
// Double-buffered LDS filled with global_load_async_to_lds_b128.
// ---------------------------------------------------------------------------
struct GemmArgs {
  const unsigned short* A;   // bf16 [M,Kd] row-major (k fastest)
  const unsigned short* B;   // bf16 [N,Kd] row-major (k fastest)
  float*          C_f32;     // optional outputs
  unsigned short* C_bf;
  unsigned short* C2_bf;     // bf16, scaled per-column by colscale
  unsigned short* Ct_bf;     // bf16, transposed per 256-row block
  const float* bias_vec;     // [N] or null
  const float* bias_scalar;  // device scalar or null
  const float* colscale;     // [N] or null
  int M, N, Kd;
  long long sA, sB, sC;      // per-batch element strides
  int relu;
};

#define BM 64
#define BN 128
#define BK 32
#define PITCH 20               // dwords per LDS row (80 B: b128-aligned, no bank conflicts)

__global__ __launch_bounds__(256) void wmma_gemm_kernel(GemmArgs g) {
  __shared__ __align__(16) unsigned int lds_a[2][BM * PITCH];
  __shared__ __align__(16) unsigned int lds_b[2][BN * PITCH];

  const int tid  = threadIdx.x;
  const int lane = tid & 31;
  const int wave = tid >> 5;
  const int wm   = wave >> 2;       // 0..1 (M)
  const int wn   = wave & 3;        // 0..3 (N)
  const int hi   = lane >> 4;
  const int l16  = lane & 15;

  const int n0 = blockIdx.x * BN;
  const int m0 = blockIdx.y * BM;
  const long long batch = blockIdx.z;

  const unsigned short* A = g.A + batch * g.sA;
  const unsigned short* B = g.B + batch * g.sB;

  // Per-thread async chunk assignment (16B chunks; 64B of bf16 per tile row).
  const int a_row = tid >> 2, a_c = tid & 3;                 // A: 1 chunk/thread
  int a_rowg = m0 + a_row; if (a_rowg > g.M - 1) a_rowg = g.M - 1;
  const unsigned short* a_src = A + (long long)a_rowg * g.Kd + a_c * 8;
  const int b_row0 = tid >> 2;                               // B rows 0..63
  const int b_row1 = (tid + 256) >> 2;                       // B rows 64..127
  const unsigned short* b_src0 = B + (long long)(n0 + b_row0) * g.Kd + a_c * 8;
  const unsigned short* b_src1 = B + (long long)(n0 + b_row1) * g.Kd + a_c * 8;

  unsigned int a_dst[2], b_dst0[2], b_dst1[2];
  for (int z = 0; z < 2; z++) {
    a_dst[z]  = lds_off_of(&lds_a[z][0]) + (a_row  * PITCH + a_c * 4) * 4;
    b_dst0[z] = lds_off_of(&lds_b[z][0]) + (b_row0 * PITCH + a_c * 4) * 4;
    b_dst1[z] = lds_off_of(&lds_b[z][0]) + (b_row1 * PITCH + a_c * 4) * 4;
  }

  auto issue = [&](int z, int k0) {
    async_copy_b128(a_dst[z],  a_src  + k0);
    async_copy_b128(b_dst0[z], b_src0 + k0);
    async_copy_b128(b_dst1[z], b_src1 + k0);
  };

  v8f_t acc[2][2];
  for (int i = 0; i < 2; i++)
    for (int j = 0; j < 2; j++)
      for (int e = 0; e < 8; e++) acc[i][j][e] = 0.0f;

  issue(0, 0);
  int buf = 0;
  for (int k0 = 0; k0 < g.Kd; k0 += BK) {
    if (k0 + BK < g.Kd) { issue(buf ^ 1, k0 + BK); wait_async<3>(); }
    else                { wait_async<0>(); }
    __syncthreads();

    union Frag { v16bf_t v; unsigned int u[8]; } af[2], bmat[2];
    for (int mi = 0; mi < 2; mi++) {
      int mrow = wm * 32 + mi * 16 + l16;
      for (int r = 0; r < 8; r++) {
        // A 16x32 bf16: lanes0-15 K=0..7(v0-3),16..23(v4-7); lanes16-31 +8
        int kidx = (r < 4) ? (2 * r + 8 * hi) : (16 + 2 * (r - 4) + 8 * hi);
        af[mi].u[r] = lds_a[buf][mrow * PITCH + (kidx >> 1)];
      }
    }
    for (int ni = 0; ni < 2; ni++) {
      int ncol = wn * 32 + ni * 16 + l16;
      for (int r = 0; r < 8; r++)   // B 32x16: lanes0-15 K=0..7,16..23; lanes16-31 +8
        bmat[ni].u[r] = lds_b[buf][ncol * PITCH + ((r < 4) ? r : r + 4) + 4 * hi];
    }
    for (int mi = 0; mi < 2; mi++)
      for (int ni = 0; ni < 2; ni++)
        acc[mi][ni] = __builtin_amdgcn_wmma_f32_16x16x32_bf16(
            false, af[mi].v, false, bmat[ni].v, (short)0, acc[mi][ni],
            false, false);
    __syncthreads();
    buf ^= 1;
  }

  // ---- epilogue ----
  const float biasS = g.bias_scalar ? *g.bias_scalar : 0.0f;
  for (int mi = 0; mi < 2; mi++) {
    for (int ni = 0; ni < 2; ni++) {
      int ncol = n0 + wn * 32 + ni * 16 + l16;
      float bv = g.bias_vec ? g.bias_vec[ncol] : biasS;
      float cs = g.colscale ? g.colscale[ncol] : 1.0f;
      for (int r = 0; r < 8; r++) {
        int mrow = m0 + wm * 32 + mi * 16 + r + 8 * hi;   // C: m = vgpr + 8*hi
        if (mrow < g.M) {
          float v = acc[mi][ni][r] + bv;
          if (g.relu && v < 0.0f) v = 0.0f;
          long long o = batch * g.sC + (long long)mrow * g.N + ncol;
          if (g.C_f32) g.C_f32[o] = v;
          if (g.C_bf)  g.C_bf[o]  = f2bf(v);
          if (g.C2_bf) g.C2_bf[o] = f2bf(v * cs);
          if (g.Ct_bf)                                     // [blk][ncol][row%256]
            g.Ct_bf[(long long)(mrow >> 8) * g.N * 256 +
                    (long long)ncol * 256 + (mrow & 255)] = f2bf(v);
        }
      }
    }
  }
}

// ---------------------------------------------------------------------------
// Elementwise / reduction helper kernels
// ---------------------------------------------------------------------------
__global__ __launch_bounds__(256) void cvt_bf16_kernel(
    const float* src, unsigned short* dst, long long n, const float* scale_ptr) {
  float s = scale_ptr ? *scale_ptr : 1.0f;
  long long i0 = ((long long)blockIdx.x * 256 + threadIdx.x) * 8;
  if (i0 + 8 <= n) {
    float4 a = *(const float4*)(src + i0);
    float4 b = *(const float4*)(src + i0 + 4);
    uint4 o;
    o.x = pack2(a.x * s, a.y * s); o.y = pack2(a.z * s, a.w * s);
    o.z = pack2(b.x * s, b.y * s); o.w = pack2(b.z * s, b.w * s);
    *(uint4*)(dst + i0) = o;
  } else {
    for (long long i = i0; i < n; i++) dst[i] = f2bf(src[i] * s);
  }
}

__global__ void normsq_kernel(const float* a, const float* b, const float* c,
                              float* out) {
  __shared__ float red[256];
  const float* src; long long n;
  if (blockIdx.x == 0)      { src = a; n = 768LL * 1024; }
  else if (blockIdx.x == 1) { src = b; n = 768LL * 1024; }
  else                      { src = c; n = 8LL * 768; }
  float s = 0.0f;
  for (long long i = threadIdx.x; i < n; i += 256) { float x = src[i]; s += x * x; }
  red[threadIdx.x] = s; __syncthreads();
  for (int o = 128; o > 0; o >>= 1) {
    if (threadIdx.x < o) red[threadIdx.x] += red[threadIdx.x + o];
    __syncthreads();
  }
  if (threadIdx.x == 0) out[blockIdx.x] = red[0];
}

__global__ void prep_kernel(const float* nsq, const float* v_g, const float* q_g,
                            const float* h_g, const float* h_mat_v,
                            const float* h_bias, float* scales, float* hsum) {
  int t = threadIdx.x;                          // 768 threads
  float sh = h_g[0] * rsqrtf(nsq[2]);
  float s = 0.0f;
  for (int h = 0; h < 8; h++) s += h_mat_v[h * 768 + t];
  hsum[t] = s * sh;
  if (t == 0) {
    scales[0] = v_g[0] * rsqrtf(nsq[0]);        // sv = g/||v_w||
    scales[1] = q_g[0] * rsqrtf(nsq[1]);        // sq
    float bs = 0.0f;
    for (int h = 0; h < 8; h++) bs += h_bias[h];
    scales[2] = bs;                             // summed h_bias
  }
}

__global__ __launch_bounds__(256) void fusion_kernel(
    const unsigned short* v_bf, const unsigned short* tmp_bf, float* logits) {
  int b = blockIdx.x, t = threadIdx.x;          // t = h-channel 0..255
  float acc = 0.0f;
  for (int v = 0; v < 256; v++) {
    long long base = ((long long)(b * 256 + v)) * 768 + 3 * t;
    acc += bf2f(v_bf[base + 0]) * bf2f(tmp_bf[base + 0])
         + bf2f(v_bf[base + 1]) * bf2f(tmp_bf[base + 1])
         + bf2f(v_bf[base + 2]) * bf2f(tmp_bf[base + 2]);
  }
  logits[b * 256 + t] = acc;                    // mean(K)*K == group sum
}

__global__ void bn_kernel(const float* logits, const float* gamma,
                          const float* beta, unsigned short* out_bf) {
  int h = threadIdx.x;                          // 256 threads
  float x[16], s = 0.0f, s2 = 0.0f;
  for (int b = 0; b < 16; b++) {
    x[b] = logits[b * 256 + h]; s += x[b]; s2 += x[b] * x[b];
  }
  float mean = s * (1.0f / 16.0f);
  float var  = s2 * (1.0f / 16.0f) - mean * mean;
  float inv  = rsqrtf(var + 1e-5f) * gamma[h];
  for (int b = 0; b < 16; b++)
    out_bf[b * 256 + h] = f2bf((x[b] - mean) * inv + beta[h]);
}

// ---------------------------------------------------------------------------
static void launch_gemm(hipStream_t stream, const GemmArgs& g, int batches) {
  dim3 grid(g.N / BN, (g.M + BM - 1) / BM, batches);
  wmma_gemm_kernel<<<grid, dim3(256), 0, stream>>>(g);
}
static void launch_cvt(hipStream_t stream, const float* src, unsigned short* dst,
                       long long n, const float* scale) {
  long long blocks = (n + 2047) / 2048;
  cvt_bf16_kernel<<<dim3((unsigned)blocks), dim3(256), 0, stream>>>(src, dst, n, scale);
}

extern "C" void kernel_launch(void* const* d_in, const int* in_sizes, int n_in,
                              void* d_out, int out_size, void* d_ws, size_t ws_size,
                              hipStream_t stream) {
  const float* ab      = (const float*)d_in[0];
  const float* ag      = (const float*)d_in[1];
  const float* v_w     = (const float*)d_in[2];
  const float* v_g     = (const float*)d_in[3];
  const float* v_b     = (const float*)d_in[4];
  const float* q_w     = (const float*)d_in[5];
  const float* q_g     = (const float*)d_in[6];
  const float* q_b     = (const float*)d_in[7];
  const float* h_mat_v = (const float*)d_in[8];
  const float* h_g     = (const float*)d_in[9];
  const float* h_bias  = (const float*)d_in[10];
  const float* bn_gamma= (const float*)d_in[11];
  const float* bn_beta = (const float*)d_in[12];
  const float* fc1_w   = (const float*)d_in[13];
  const float* fc1_b   = (const float*)d_in[14];
  const float* fc2_w   = (const float*)d_in[15];
  const float* fc2_b   = (const float*)d_in[16];
  const float* out_w   = (const float*)d_in[17];
  const float* out_b   = (const float*)d_in[18];

  char* ws = (char*)d_ws;
  size_t off = 0;
  auto alloc = [&](size_t bytes) -> void* {
    void* p = ws + off;
    off = (off + bytes + 255) & ~(size_t)255;
    return p;
  };
  float* nsq              = (float*)alloc(3 * 4);
  float* scales           = (float*)alloc(3 * 4);
  float* hsum             = (float*)alloc(768 * 4);
  unsigned short* ab_bf   = (unsigned short*)alloc(4096LL * 1024 * 2);
  unsigned short* ag_bf   = (unsigned short*)alloc(4096LL * 1024 * 2);
  unsigned short* vw_bf   = (unsigned short*)alloc(768LL * 1024 * 2);
  unsigned short* qw_bf   = (unsigned short*)alloc(768LL * 1024 * 2);
  unsigned short* fc1w_bf = (unsigned short*)alloc(1024LL * 256 * 2);
  unsigned short* fc2w_bf = (unsigned short*)alloc(256LL * 1024 * 2);
  unsigned short* outw_bf = (unsigned short*)alloc(128LL * 256 * 2);
  unsigned short* v_bf    = (unsigned short*)alloc(4096LL * 768 * 2);
  unsigned short* vh_bf   = (unsigned short*)alloc(4096LL * 768 * 2);
  unsigned short* q_bf    = (unsigned short*)alloc(4096LL * 768 * 2);
  unsigned short* qT_bf   = (unsigned short*)alloc(16LL * 768 * 256 * 2);
  unsigned short* att_bf  = (unsigned short*)alloc(16LL * 256 * 256 * 2);
  unsigned short* tmp_bf  = (unsigned short*)alloc(16LL * 256 * 768 * 2);
  float* logits           = (float*)alloc(16 * 256 * 4);
  unsigned short* lgn_bf  = (unsigned short*)alloc(16 * 256 * 2);
  unsigned short* x1_bf   = (unsigned short*)alloc(16 * 1024 * 2);
  unsigned short* x2_bf   = (unsigned short*)alloc(16 * 256 * 2);

  // scalars (weight-norm) + head-summed h_mat
  normsq_kernel<<<dim3(3), dim3(256), 0, stream>>>(v_w, q_w, h_mat_v, nsq);
  prep_kernel<<<dim3(1), dim3(768), 0, stream>>>(nsq, v_g, q_g, h_g, h_mat_v,
                                                 h_bias, scales, hsum);

  // one-time bf16 conversions (weight-norm scales folded into weights)
  launch_cvt(stream, ab,    ab_bf,   4096LL * 1024, nullptr);
  launch_cvt(stream, ag,    ag_bf,   4096LL * 1024, nullptr);
  launch_cvt(stream, v_w,   vw_bf,   768LL * 1024,  scales + 0);
  launch_cvt(stream, q_w,   qw_bf,   768LL * 1024,  scales + 1);
  launch_cvt(stream, fc1_w, fc1w_bf, 1024LL * 256,  nullptr);
  launch_cvt(stream, fc2_w, fc2w_bf, 256LL * 1024,  nullptr);
  launch_cvt(stream, out_w, outw_bf, 128LL * 256,   nullptr);

  // v_ = relu(ab @ vw^T + v_b) -> v_bf, and vh_bf = v_*hsum
  GemmArgs gv = {};
  gv.A = ab_bf; gv.B = vw_bf; gv.C_bf = v_bf; gv.C2_bf = vh_bf;
  gv.bias_vec = v_b; gv.colscale = hsum;
  gv.M = 4096; gv.N = 768; gv.Kd = 1024; gv.relu = 1;
  launch_gemm(stream, gv, 1);

  // q_ = relu(ag @ qw^T + q_b) -> q_bf and transposed qT_bf
  GemmArgs gq = {};
  gq.A = ag_bf; gq.B = qw_bf; gq.C_bf = q_bf; gq.Ct_bf = qT_bf;
  gq.bias_vec = q_b;
  gq.M = 4096; gq.N = 768; gq.Kd = 1024; gq.relu = 1;
  launch_gemm(stream, gq, 1);

  // att_sum[b] = vh[b] @ q_[b]^T + sum(h_bias)
  GemmArgs ga = {};
  ga.A = vh_bf; ga.B = q_bf; ga.C_bf = att_bf; ga.bias_scalar = scales + 2;
  ga.M = 256; ga.N = 256; ga.Kd = 768;
  ga.sA = 256LL * 768; ga.sB = 256LL * 768; ga.sC = 256LL * 256;
  launch_gemm(stream, ga, 16);

  // tmp[b] = att[b] @ q_[b]  (B = qT, k-fast)
  GemmArgs gt = {};
  gt.A = att_bf; gt.B = qT_bf; gt.C_bf = tmp_bf;
  gt.M = 256; gt.N = 768; gt.Kd = 256;
  gt.sA = 256LL * 256; gt.sB = 768LL * 256; gt.sC = 256LL * 768;
  launch_gemm(stream, gt, 16);

  // fusion dot + K-group sum -> logits; batch-norm -> bf16 MLP input
  fusion_kernel<<<dim3(16), dim3(256), 0, stream>>>(v_bf, tmp_bf, logits);
  bn_kernel<<<dim3(1), dim3(256), 0, stream>>>(logits, bn_gamma, bn_beta, lgn_bf);

  // MLP head (M=16 => single WMMA tile row)
  GemmArgs g1 = {};
  g1.A = lgn_bf; g1.B = fc1w_bf; g1.C_bf = x1_bf; g1.bias_vec = fc1_b;
  g1.M = 16; g1.N = 1024; g1.Kd = 256; g1.relu = 1;
  launch_gemm(stream, g1, 1);

  GemmArgs g2 = {};
  g2.A = x1_bf; g2.B = fc2w_bf; g2.C_bf = x2_bf; g2.bias_vec = fc2_b;
  g2.M = 16; g2.N = 256; g2.Kd = 1024; g2.relu = 1;
  launch_gemm(stream, g2, 1);

  GemmArgs g3 = {};
  g3.A = x2_bf; g3.B = outw_bf; g3.C_f32 = (float*)d_out; g3.bias_vec = out_b;
  g3.M = 16; g3.N = 128; g3.Kd = 256;
  launch_gemm(stream, g3, 1);
}
